// MultiHeadSelfAttention_9474697855584
// MI455X (gfx1250) — matrix operand order repaired and checked by reading, not compiled, not run on hardware
//
#include <hip/hip_runtime.h>

// ---------------------------------------------------------------------------
// MI455X (gfx1250) fused MHSA: bf16 WMMA GEMMs + flash attention, f32 accum.
// Uses CDNA5 async global->LDS copies (ASYNCcnt) for pure-copy staging.
// ---------------------------------------------------------------------------

typedef __attribute__((ext_vector_type(16))) __bf16 v16bf;
typedef __attribute__((ext_vector_type(8)))  float  v8f;

constexpr int Bb  = 4;
constexpr int Ss  = 2048;
constexpr int Dd  = 1024;
constexpr int Hh  = 16;
constexpr int HDd = 64;
constexpr int Mtot = Bb * Ss;   // 8192

__device__ inline __bf16 f2bf(float f) { return (__bf16)f; }  // native v_cvt

// A fragment: 16x32 bf16 tile, element (m,k) = p[m*stride + k].
// Lanes 0-15 hold row m=lane (K 0..15); lanes 16-31 same rows (K 16..31).
__device__ inline v16bf frag_a(const __bf16* p, int stride) {
  const int lane = threadIdx.x & 31;
  const int m  = lane & 15;
  const int kh = (lane >> 4) * 16;
  v16bf a;
#pragma unroll
  for (int j = 0; j < 8; ++j) {
    const int kb = 2 * j + kh;
    a[2 * j]     = p[m * stride + kb];
    a[2 * j + 1] = p[m * stride + kb + 1];
  }
  return a;
}

// B fragment, source laid out [n][k]: element (k,n) = p[n*stride + k].
__device__ inline v16bf frag_b_nk(const __bf16* p, int stride) {
  const int lane = threadIdx.x & 31;
  const int n  = lane & 15;
  const int kh = (lane >> 4) * 16;
  v16bf b;
#pragma unroll
  for (int j = 0; j < 8; ++j) {
    b[2 * j]     = p[n * stride + kh + 2 * j];
    b[2 * j + 1] = p[n * stride + kh + 2 * j + 1];
  }
  return b;
}

// B fragment, source laid out [k][n]: element (k,n) = p[k*stride + n].
__device__ inline v16bf frag_b_kn(const __bf16* p, int stride) {
  const int lane = threadIdx.x & 31;
  const int n  = lane & 15;
  const int kh = (lane >> 4) * 16;
  v16bf b;
#pragma unroll
  for (int j = 0; j < 8; ++j) {
    b[2 * j]     = p[(kh + 2 * j) * stride + n];
    b[2 * j + 1] = p[(kh + 2 * j + 1) * stride + n];
  }
  return b;
}

__device__ inline v8f wmma_bf16(v16bf a, v16bf b, v8f c) {
  return __builtin_amdgcn_wmma_f32_16x16x32_bf16(false, a, false, b, (short)0, c,
                                                 false, false);
}

// async global->LDS copy helpers (CDNA5; tracked on ASYNCcnt)
__device__ inline void async_copy_b128(void* lds_dst, const void* gsrc) {
  unsigned loff = (unsigned)(size_t)lds_dst;
  unsigned long long ga = (unsigned long long)(size_t)gsrc;
  asm volatile("global_load_async_to_lds_b128 %0, %1, off"
               :: "v"(loff), "v"(ga) : "memory");
}
__device__ inline void async_copy_b64(void* lds_dst, const void* gsrc) {
  unsigned loff = (unsigned)(size_t)lds_dst;
  unsigned long long ga = (unsigned long long)(size_t)gsrc;
  asm volatile("global_load_async_to_lds_b64 %0, %1, off"
               :: "v"(loff), "v"(ga) : "memory");
}
__device__ inline void async_wait0() {
  asm volatile("s_wait_asynccnt 0x0" ::: "memory");
}

// ---------------------------------------------------------------------------
// GEMM: C[M,N] = A[M,K] @ W[N,K]^T   (M=8192, N=K=1024)
//  A_BF16=false : A is f32 (x), converted to bf16 into LDS
//  A_BF16=true  : A is bf16, staged via async global->LDS copies
//  OUT_PERMUTE=true : store bf16 to [B,H,S,HD] (QKV path); else f32 row-major
// Block tile 128x128x32, 256 threads = 8 waves (4x2 wave grid, 32x64 per wave)
// ---------------------------------------------------------------------------
template <bool A_BF16, bool OUT_PERMUTE>
__global__ __launch_bounds__(256) void gemm_bf16_wmma(const void* __restrict__ Ap,
                                                      const float* __restrict__ W,
                                                      void* __restrict__ Cp) {
  constexpr int BM = 128, BN = 128, BK = 32, LS = BK + 4;  // 72B row pitch
  __shared__ __bf16 ldsA[BM * LS];
  __shared__ __bf16 ldsB[BN * LS];

  const int tid  = threadIdx.x;
  const int wave = tid >> 5;
  const int lane = tid & 31;
  const int m0 = blockIdx.x * BM;
  const int n0 = blockIdx.y * BN;
  const int wm = wave >> 1;  // 0..3
  const int wn = wave & 1;   // 0..1

  const v8f vzero = {0.f, 0.f, 0.f, 0.f, 0.f, 0.f, 0.f, 0.f};
  v8f acc[2][4];
#pragma unroll
  for (int i = 0; i < 2; ++i)
#pragma unroll
    for (int j = 0; j < 4; ++j) acc[i][j] = vzero;

  for (int k0 = 0; k0 < Dd; k0 += BK) {
#pragma unroll
    for (int it = 0; it < 4; ++it) {
      const int id  = tid + it * 256;  // 0..1023
      const int row = id >> 3;         // 0..127
      const int c4  = (id & 7) * 4;    // 0,4,..,28
      if (A_BF16) {
        const __bf16* gA = (const __bf16*)Ap + (size_t)(m0 + row) * Dd + k0 + c4;
        async_copy_b64(&ldsA[row * LS + c4], gA);   // pure copy: async path
      } else {
        const float4 va =
            *(const float4*)((const float*)Ap + (size_t)(m0 + row) * Dd + k0 + c4);
        ldsA[row * LS + c4 + 0] = f2bf(va.x);
        ldsA[row * LS + c4 + 1] = f2bf(va.y);
        ldsA[row * LS + c4 + 2] = f2bf(va.z);
        ldsA[row * LS + c4 + 3] = f2bf(va.w);
      }
      const float4 vb = *(const float4*)(W + (size_t)(n0 + row) * Dd + k0 + c4);
      ldsB[row * LS + c4 + 0] = f2bf(vb.x);
      ldsB[row * LS + c4 + 1] = f2bf(vb.y);
      ldsB[row * LS + c4 + 2] = f2bf(vb.z);
      ldsB[row * LS + c4 + 3] = f2bf(vb.w);
      // prefetch next K-slab into L2 while we compute this one
      if (k0 + BK < Dd) {
        if (!A_BF16)
          __builtin_prefetch((const float*)Ap + (size_t)(m0 + row) * Dd + k0 + BK + c4, 0, 1);
        __builtin_prefetch(W + (size_t)(n0 + row) * Dd + k0 + BK + c4, 0, 1);
      }
    }
    if (A_BF16) async_wait0();
    __syncthreads();

    const v16bf a0 = frag_a(ldsA + (wm * 32) * LS, LS);
    const v16bf a1 = frag_a(ldsA + (wm * 32 + 16) * LS, LS);
#pragma unroll
    for (int j = 0; j < 4; ++j) {
      const v16bf bf = frag_b_nk(ldsB + (wn * 64 + j * 16) * LS, LS);
      acc[0][j] = wmma_bf16(a0, bf, acc[0][j]);
      acc[1][j] = wmma_bf16(a1, bf, acc[1][j]);
    }
    __syncthreads();
  }

  // C fragment layout: lanes 0-15 -> rows r, cols lane; lanes 16-31 -> rows 8+r
  const int colc = lane & 15;
  const int rowb = (lane >> 4) * 8;
#pragma unroll
  for (int i = 0; i < 2; ++i)
#pragma unroll
    for (int j = 0; j < 4; ++j)
#pragma unroll
      for (int r = 0; r < 8; ++r) {
        const int m = m0 + wm * 32 + i * 16 + rowb + r;
        const int n = n0 + wn * 64 + j * 16 + colc;
        const float v = acc[i][j][r];
        if (OUT_PERMUTE) {
          const int bb = m >> 11;        // m / S
          const int s  = m & (Ss - 1);   // m % S
          const int hh = n >> 6;         // n / HD
          const int hd = n & 63;
          ((__bf16*)Cp)[(((size_t)bb * Hh + hh) * Ss + s) * HDd + hd] = f2bf(v);
        } else {
          ((float*)Cp)[(size_t)m * Dd + n] = v;
        }
      }
}

// ---------------------------------------------------------------------------
// Flash attention (causal). Q/K/V bf16 [B,H,S,HD]. Block = 8 waves; each wave
// owns a 16-row query strip (128 rows per block per (b,h)). K/V tiles staged
// once per block into LDS via async copies (8x less global traffic). Online
// softmax in f32; P converted to bf16 via LDS transpose for the PV WMMA.
// ---------------------------------------------------------------------------
__global__ __launch_bounds__(256) void flash_attn_wmma(const __bf16* __restrict__ Q,
                                                       const __bf16* __restrict__ K,
                                                       const __bf16* __restrict__ V,
                                                       __bf16* __restrict__ Oa) {
  constexpr int KVS = 72;                  // K/V LDS row pitch (144B, 16B aligned)
  __shared__ __bf16 ldsK[64 * KVS];        // 64 keys x 64 hd
  __shared__ __bf16 ldsV[64 * KVS];
  __shared__ __bf16 ldsP[8][16 * KVS];     // per-wave 16x64 P tile

  const int tid  = threadIdx.x;
  const int wave = tid >> 5;
  const int lane = tid & 31;
  const int bh = blockIdx.y;
  const int bq = bh >> 4;   // batch
  const int hq = bh & 15;   // head
  const int q0 = blockIdx.x * 128 + wave * 16;

  const __bf16* Qb = Q + (size_t)bh * Ss * HDd;
  const __bf16* Kb = K + (size_t)bh * Ss * HDd;
  const __bf16* Vb = V + (size_t)bh * Ss * HDd;

  const v16bf qa0 = frag_a(Qb + (size_t)q0 * HDd, HDd);
  const v16bf qa1 = frag_a(Qb + (size_t)q0 * HDd + 32, HDd);

  const v8f vzero = {0.f, 0.f, 0.f, 0.f, 0.f, 0.f, 0.f, 0.f};
  v8f o[4];
#pragma unroll
  for (int j = 0; j < 4; ++j) o[j] = vzero;

  float mrow[8], lrow[8];
#pragma unroll
  for (int r = 0; r < 8; ++r) { mrow[r] = -1e30f; lrow[r] = 0.f; }

  const int colc = lane & 15;
  const int rowb = (lane >> 4) * 8;
  const int ktmax = blockIdx.x * 128 + 64;  // block-uniform loop bound

  for (int kt0 = 0; kt0 <= ktmax; kt0 += 64) {
    // --- stage 64x64 K and V tiles into LDS (async, block-wide) ---
    {
      const __bf16* gK = Kb + (size_t)kt0 * HDd;
      const __bf16* gV = Vb + (size_t)kt0 * HDd;
#pragma unroll
      for (int it = 0; it < 4; ++it) {
        const int c = tid + it * 256;     // 0..1023 (16B chunks)
        const int tsel = c >> 9;          // 0:K  1:V
        const int idx = c & 511;
        const int row = idx >> 3;         // 0..63
        const int e8  = (idx & 7) * 8;    // element offset (8 bf16 = 16B)
        const __bf16* g = (tsel ? gV : gK) + (size_t)row * HDd + e8;
        __bf16* l = (tsel ? ldsV : ldsK) + row * KVS + e8;
        async_copy_b128(l, g);
      }
      async_wait0();
    }
    __syncthreads();

    const bool active = (kt0 <= q0 + 15);
    if (active) {
      v8f s[4];
#pragma unroll
      for (int j = 0; j < 4; ++j) s[j] = vzero;

      // S = Q @ K^T  (16x64 scores; K-dim = HD = 64 -> two WMMA per strip)
#pragma unroll
      for (int j = 0; j < 4; ++j) {
        const v16bf kb0 = frag_b_nk(ldsK + (j * 16) * KVS, KVS);
        s[j] = wmma_bf16(qa0, kb0, s[j]);
        const v16bf kb1 = frag_b_nk(ldsK + (j * 16) * KVS + 32, KVS);
        s[j] = wmma_bf16(qa1, kb1, s[j]);
      }

      // scale 1/sqrt(HD) + causal mask (only diagonal-crossing tiles)
      const bool diag = (kt0 + 63 > q0);
#pragma unroll
      for (int j = 0; j < 4; ++j)
#pragma unroll
        for (int r = 0; r < 8; ++r) {
          float v = s[j][r] * 0.125f;
          if (diag) {
            const int cc = kt0 + j * 16 + colc;
            const int rr = q0 + rowb + r;
            if (cc > rr) v = -1e30f;
          }
          s[j][r] = v;
        }

      // online softmax: row max over 4 strips + 16 lanes
      float sf[8];
#pragma unroll
      for (int r = 0; r < 8; ++r) {
        float mx = fmaxf(fmaxf(s[0][r], s[1][r]), fmaxf(s[2][r], s[3][r]));
#pragma unroll
        for (int off = 1; off < 16; off <<= 1) mx = fmaxf(mx, __shfl_xor(mx, off, 32));
        const float nm = fmaxf(mrow[r], mx);
        sf[r] = __expf(mrow[r] - nm);
        mrow[r] = nm;
      }

      float rs[8];
#pragma unroll
      for (int r = 0; r < 8; ++r) rs[r] = 0.f;
#pragma unroll
      for (int j = 0; j < 4; ++j)
#pragma unroll
        for (int r = 0; r < 8; ++r) {
          const float p = __expf(s[j][r] - mrow[r]);
          rs[r] += p;
          ldsP[wave][(rowb + r) * KVS + j * 16 + colc] = f2bf(p);
        }
#pragma unroll
      for (int r = 0; r < 8; ++r) {
#pragma unroll
        for (int off = 1; off < 16; off <<= 1) rs[r] += __shfl_xor(rs[r], off, 32);
        lrow[r] = lrow[r] * sf[r] + rs[r];
      }
      // rescale running O
#pragma unroll
      for (int j = 0; j < 4; ++j)
#pragma unroll
        for (int r = 0; r < 8; ++r) o[j][r] *= sf[r];
    }
    __syncthreads();  // P (C-layout) visible before A-fragment reload
    if (active) {
      const v16bf pa0 = frag_a(&ldsP[wave][0], KVS);
      const v16bf pa1 = frag_a(&ldsP[wave][32], KVS);
#pragma unroll
      for (int j = 0; j < 4; ++j) {
        const v16bf vb0 = frag_b_kn(ldsV + j * 16, KVS);
        o[j] = wmma_bf16(pa0, vb0, o[j]);
        const v16bf vb1 = frag_b_kn(ldsV + 32 * KVS + j * 16, KVS);
        o[j] = wmma_bf16(pa1, vb1, o[j]);
      }
    }
    __syncthreads();  // protect ldsK/ldsV/ldsP before next iteration's stage
  }

  // epilogue: normalize and write attn output as [B,S,D] bf16 workspace
#pragma unroll
  for (int r = 0; r < 8; ++r) {
    const float inv = __builtin_amdgcn_rcpf(lrow[r]);
    const int srow = q0 + rowb + r;
#pragma unroll
    for (int j = 0; j < 4; ++j) {
      const float v = o[j][r] * inv;
      const int n = hq * HDd + j * 16 + colc;
      Oa[((size_t)bq * Ss + srow) * Dd + n] = f2bf(v);
    }
  }
}

// ---------------------------------------------------------------------------
extern "C" void kernel_launch(void* const* d_in, const int* in_sizes, int n_in,
                              void* d_out, int out_size, void* d_ws, size_t ws_size,
                              hipStream_t stream) {
  (void)in_sizes; (void)n_in; (void)out_size; (void)ws_size;
  const float* x  = (const float*)d_in[0];
  const float* Wq = (const float*)d_in[1];
  const float* Wk = (const float*)d_in[2];
  const float* Wv = (const float*)d_in[3];
  const float* Wo = (const float*)d_in[4];

  const size_t per = (size_t)Bb * Ss * Dd;  // elements per tensor
  __bf16* qws = (__bf16*)d_ws;              // [B,H,S,HD] bf16
  __bf16* kws = qws + per;
  __bf16* vws = kws + per;
  __bf16* aws = vws + per;                  // attn out [B,S,D] bf16

  dim3 blk(256);
  dim3 gproj(Mtot / 128, Dd / 128);  // (64, 8)

  gemm_bf16_wmma<false, true><<<gproj, blk, 0, stream>>>((const void*)x, Wq, (void*)qws);
  gemm_bf16_wmma<false, true><<<gproj, blk, 0, stream>>>((const void*)x, Wk, (void*)kws);
  gemm_bf16_wmma<false, true><<<gproj, blk, 0, stream>>>((const void*)x, Wv, (void*)vws);

  flash_attn_wmma<<<dim3(Ss / 128, Bb * Hh), blk, 0, stream>>>(qws, kws, vws, aws);

  gemm_bf16_wmma<true, false><<<gproj, blk, 0, stream>>>((const void*)aws, Wo, d_out);
}
